// SteeredConv2d_78606491452290
// MI455X (gfx1250) — matrix operand order repaired and checked by reading, not compiled, not run on hardware
//
#include <hip/hip_runtime.h>

// Steerable 3x3 conv, refactored as (1x1 channel GEMM via WMMA f16) o (fixed
// 3x3 basis stencil + per-pixel harmonic fold), fully fused in LDS.
//
//   y_k[o,p] = sum_i w_k[o,i] * x[i,p]            (k = 0..3, M=256 GEMM, K=64)
//   out[o,p] = y0[p] + (1/sqrt8)*sum_ring y1
//            + sum_t (c1*cos(ang_t)+s1*sin(ang_t))/2 * y2[p+t]
//            + sum_t (c2*cos2+s2*sin2)/2           * y3[p+t]  + bias[o]
//
// Per workgroup: one 16x16 output tile (18x18 halo), one image. LDS holds the
// f16 weights (32KB), x halo tile (42KB) and the full y intermediate stored
// pixel-major / channel-contiguous with a 264-element padded row (173KB):
// 253KB of the WGP's 320KB, so y never touches HBM. The channel-contiguous
// layout lets Phase B store each D fragment as one ds_store_b128 (lane holds 8
// consecutive M rows) and Phase C fetch 8 channels per tap with one
// ds_load_b128 (8x fewer LDS issues than scalar u16). Row pad 256->264 f16
// (528B) breaks the 512B bank aliasing. HBM traffic = x + theta + out ~135MB
// -> ~6us floor at 23.3TB/s; WMMA work ~11 GFLOP f16 (vs 96.6 GFLOP naive).

typedef __attribute__((ext_vector_type(16))) _Float16 v16h;
typedef __attribute__((ext_vector_type(8)))  _Float16 v8h;
typedef __attribute__((ext_vector_type(8)))  float    v8f;

#define TILE 16
#define HALO 18            // TILE + 2
#define NPIX (HALO * HALO) // 324
#define NPIX_PAD 336       // 21 * 16 (padded to whole N-tiles)
#define NTILES_N 21
#define CIN 64
#define COUT 64
#define MROWS 256          // 4 weight sets * 64 out channels
#define YSTRIDE 264        // 256 + 8 f16 pad: 528B row, kills 512B bank alias
#define IMG_H 128
#define IMG_W 128

#define LDS_AW (MROWS * CIN)         // 16384 f16 (32768 B)
#define LDS_XS (NPIX_PAD * CIN)      // 21504 f16 (43008 B)
#define LDS_YS (NPIX_PAD * YSTRIDE)  // 88704 f16 (177408 B)
#define LDS_TOTAL (LDS_AW + LDS_XS + LDS_YS)   // 253,184 B < 320 KB

__global__ __launch_bounds__(256)
void steered_conv2d_fused(const float* __restrict__ x,
                          const float* __restrict__ theta,
                          const float* __restrict__ wgt,   // (O, I, 4)
                          const float* __restrict__ bias,  // (O)
                          float* __restrict__ out)
{
    extern __shared__ _Float16 smem[];
    _Float16* Aw = smem;                 // [c'=k*64+o][i]
    _Float16* xs = Aw + LDS_AW;          // [p][i]    p = halo pixel
    _Float16* ys = xs + LDS_XS;          // [p][c']   channel-contiguous

    const int tid = threadIdx.x;
    const int tx = blockIdx.x, ty = blockIdx.y, n = blockIdx.z;
    const int gx0 = tx * TILE, gy0 = ty * TILE;

    // ---- Phase A1: stage steerable weights as f16, Aw[(k*64+o)*64 + i] ----
    for (int idx = tid; idx < MROWS * CIN; idx += 256) {
        int cp = idx >> 6;         // c' = k*64+o
        int i  = idx & 63;
        int o  = cp & 63;
        int k  = cp >> 6;
        Aw[idx] = (_Float16)wgt[(o * CIN + i) * 4 + k];
    }

    // ---- Phase A2: stage x halo tile (SAME padding), xs[p*64 + i] ----
    const float* xn = x + (size_t)n * CIN * IMG_H * IMG_W;
    for (int idx = tid; idx < CIN * NPIX_PAD; idx += 256) {
        int i = idx / NPIX_PAD;              // channel
        int p = idx - i * NPIX_PAD;          // halo pixel (consecutive tid -> consecutive p)
        float v = 0.f;
        if (p < NPIX) {
            int hy = p / HALO, hx = p - hy * HALO;
            int gy = gy0 + hy - 1, gx = gx0 + hx - 1;
            if ((unsigned)gy < IMG_H && (unsigned)gx < IMG_W)
                v = xn[(size_t)i * IMG_H * IMG_W + gy * IMG_W + gx];
        }
        xs[p * CIN + i] = (_Float16)v;
    }
    __syncthreads();

    // ---- Phase B: y[c',p] = sum_i Aw[c',i] * xs[p,i]  (WMMA f16, f32 acc) ----
    const int wid  = tid >> 5;
    const int lane = tid & 31;
    const int lm   = lane & 15;   // M (A) / N (B,D) index
    const int hi   = lane >> 4;   // lane-half selector
    for (int t = wid; t < 16 * NTILES_N; t += 8) {      // 336 tiles, 42/wave
        int mt = t / NTILES_N;
        int nt = t - mt * NTILES_N;
        int arow = (mt * 16 + lm) * CIN;
        int brow = (nt * 16 + lm) * CIN;
        v8f c = {};
#pragma unroll
        for (int kt = 0; kt < 2; ++kt) {                // K = 64 = 2 x 32
            int kb = kt * 32;
            v16h a, b;
#pragma unroll
            for (int e = 0; e < 16; ++e) {
                // A 16x32 f16 layout: e<8 -> K=e+8*hi ; e>=8 -> K=e+8+8*hi
                int ka = kb + ((e < 8) ? (e + 8 * hi) : (e + 8 + 8 * hi));
                a[e] = Aw[arow + ka];
            }
#pragma unroll
            for (int e = 0; e < 16; ++e) {
                // B 32x16 f16 layout: e<8 -> K=e+8*hi ; e>=8 -> K=e+8+8*hi
                b[e] = xs[brow + kb + ((e < 8) ? (e + 8 * hi) : (e + 8 + 8 * hi))];
            }
            c = __builtin_amdgcn_wmma_f32_16x16x32_f16(
                    false, a, false, b, (short)0, c, false, false);
        }
        // D: lane holds pixel column n=lm, 8 consecutive rows m = 8*hi + r
        // -> pack to 8 x f16 and store with a single ds_store_b128
        v8h h;
#pragma unroll
        for (int r = 0; r < 8; ++r) h[r] = (_Float16)c[r];
        *(v8h*)(ys + (size_t)(nt * 16 + lm) * YSTRIDE + mt * 16 + hi * 8) = h;
    }
    __syncthreads();

    // ---- Phase C: 3x3 basis stencil + harmonic fold; thread <-> pixel ----
    const int py = tid >> 4, px = tid & 15;
    const int gy = gy0 + py, gx = gx0 + px;
    float th = theta[(size_t)n * IMG_H * IMG_W + gy * IMG_W + gx];
    float a1 = 6.28318530717958647692f * th;
    float s1, c1;
    __sincosf(a1, &s1, &c1);
    float c2 = 2.f * c1 * c1 - 1.f;
    float s2 = 2.f * s1 * c1;

    // ring taps t=0..7: (dx,dy) = E, SE, S, SW, W, NW, N, NE (y = row)
    const int   rdx[8] = { 1, 1, 0,-1,-1,-1, 0, 1};
    const int   rdy[8] = { 0, 1, 1, 1, 0,-1,-1,-1};
    const float ct[8]  = { 1.f, 0.70710678f, 0.f,-0.70710678f,-1.f,-0.70710678f, 0.f, 0.70710678f};
    const float st[8]  = { 0.f, 0.70710678f, 1.f, 0.70710678f, 0.f,-0.70710678f,-1.f,-0.70710678f};
    const float c2t[8] = { 1.f, 0.f,-1.f, 0.f, 1.f, 0.f,-1.f, 0.f};
    const float s2t[8] = { 0.f, 1.f, 0.f,-1.f, 0.f, 1.f, 0.f,-1.f};
    const float B1 = 0.35355339059327376f;  // ring basis 1/sqrt(8)

    float k1c[8], k2c[8];
    int   roff[8];
#pragma unroll
    for (int t = 0; t < 8; ++t) {
        k1c[t]  = 0.5f * (c1 * ct[t]  + s1 * st[t]);   // basis norm = 2
        k2c[t]  = 0.5f * (c2 * c2t[t] + s2 * s2t[t]);
        roff[t] = (py + 1 + rdy[t]) * HALO + (px + 1 + rdx[t]);
    }
    const int pc = (py + 1) * HALO + (px + 1);

    float* outn = out + (size_t)n * COUT * IMG_H * IMG_W + gy * IMG_W + gx;
    for (int og = 0; og < 8; ++og) {          // 8 channels per b128 fetch
        const int o0 = og * 8;
        float acc[8];
        v8h b0 = *(const v8h*)(ys + (size_t)pc * YSTRIDE + o0);     // y0 center
#pragma unroll
        for (int j = 0; j < 8; ++j) acc[j] = (float)b0[j] + bias[o0 + j];
#pragma unroll
        for (int t = 0; t < 8; ++t) {
            const _Float16* row = ys + (size_t)roff[t] * YSTRIDE;
            v8h v1 = *(const v8h*)(row + 64  + o0);
            v8h v2 = *(const v8h*)(row + 128 + o0);
            v8h v3 = *(const v8h*)(row + 192 + o0);
#pragma unroll
            for (int j = 0; j < 8; ++j) {
                acc[j] += B1     * (float)v1[j];
                acc[j] += k1c[t] * (float)v2[j];
                acc[j] += k2c[t] * (float)v3[j];
            }
        }
#pragma unroll
        for (int j = 0; j < 8; ++j)
            outn[(size_t)(o0 + j) * IMG_H * IMG_W] = acc[j];
    }
}

extern "C" void kernel_launch(void* const* d_in, const int* in_sizes, int n_in,
                              void* d_out, int out_size, void* d_ws, size_t ws_size,
                              hipStream_t stream) {
    const float* x     = (const float*)d_in[0];  // (16,64,128,128)
    const float* theta = (const float*)d_in[1];  // (16,1,128,128)
    const float* wgt   = (const float*)d_in[2];  // (64,64,4)
    const float* bias  = (const float*)d_in[3];  // (64)
    float* out = (float*)d_out;                  // (16,64,128,128)

    dim3 grid(IMG_W / TILE, IMG_H / TILE, 16);   // 8 x 8 x 16 workgroups
    dim3 block(256);                             // 8 wave32
    size_t smem = (size_t)LDS_TOTAL * sizeof(_Float16); // 253,184 B < 320 KB
    steered_conv2d_fused<<<grid, block, smem, stream>>>(x, theta, wgt, bias, out);
}
